// Block_9938554323537
// MI455X (gfx1250) — compile-verified, matrix operand order checked by reading, not run on hardware
//
#include <hip/hip_runtime.h>
#include <math.h>

typedef __bf16 bf16;
typedef __attribute__((ext_vector_type(16))) __bf16 v16bf;
typedef __attribute__((ext_vector_type(8)))  __bf16 v8bf;
typedef __attribute__((ext_vector_type(8)))  float  v8f;

union FragAB { v16bf v; v8bf h[2]; };

// ---------------- reductions (wave32) ----------------
__device__ __forceinline__ float wave_reduce_sum(float s) {
  s += __shfl_xor(s, 16, 32);
  s += __shfl_xor(s, 8, 32);
  s += __shfl_xor(s, 4, 32);
  s += __shfl_xor(s, 2, 32);
  s += __shfl_xor(s, 1, 32);
  return s;
}

__device__ __forceinline__ float block_reduce_sum_256(float s) {
  __shared__ float sm[8];
  int lane = threadIdx.x & 31, wid = threadIdx.x >> 5;
  s = wave_reduce_sum(s);
  if (lane == 0) sm[wid] = s;
  __syncthreads();
  float t = (threadIdx.x < 8) ? sm[threadIdx.x] : 0.f;
  if (wid == 0) {
    t += __shfl_xor(t, 4, 32);
    t += __shfl_xor(t, 2, 32);
    t += __shfl_xor(t, 1, 32);
    if (lane == 0) sm[0] = t;
  }
  __syncthreads();
  float r = sm[0];
  __syncthreads();
  return r;
}

// ---------------- WMMA GEMM: C = A(MxK,bf16) * W(NxK,bf16)^T + epilogue ----
// EPI: 1 = +bias -> bf16
//      2 = +bias + add_bf16 -> bf16
//      3 = +bias + add_f32  -> f32
//      4 = +bias, exact GELU -> bf16
template <int EPI, bool MAP_A, bool MAP_OUT>
__global__ __launch_bounds__(128) void gemm_bf16_wmma(
    const bf16* __restrict__ A, const bf16* __restrict__ W,
    const float* __restrict__ bias, const bf16* __restrict__ add_bf,
    const float* __restrict__ add_f32, bf16* __restrict__ out_bf,
    float* __restrict__ out_f32, int M, int N, int K, int rows_per, int tok_per) {
  const int lane = threadIdx.x & 31;
  const int wave = threadIdx.x >> 5;
  const int m0 = blockIdx.x * 128 + wave * 32;   // wave's 32-row strip
  const int n0 = blockIdx.y * 64;
  const int lr = lane & 15;
  const int hi = lane >> 4;
  const int koff = hi * 8;                        // lane K swizzle (16-bit A layout)

  // per-lane A row pointers (rows clamped; MAP_A gathers body tokens)
  const bf16* aptr[2];
#pragma unroll
  for (int i = 0; i < 2; ++i) {
    int m = m0 + i * 16 + lr;
    if (m > M - 1) m = M - 1;
    if (MAP_A) m = (m / rows_per) * tok_per + 1 + (m % rows_per);
    aptr[i] = A + (size_t)m * K + koff;
  }
  const bf16* bptr[4];
#pragma unroll
  for (int j = 0; j < 4; ++j) {
    int n = n0 + j * 16 + lr;
    bptr[j] = W + (size_t)n * K + koff;
  }

  v8f acc[2][4];
#pragma unroll
  for (int i = 0; i < 2; ++i)
#pragma unroll
    for (int j = 0; j < 4; ++j) {
      v8f z = {0.f, 0.f, 0.f, 0.f, 0.f, 0.f, 0.f, 0.f};
      acc[i][j] = z;
    }

  for (int kb = 0; kb < K; kb += 32) {
    FragAB a[2], b[4];
#pragma unroll
    for (int i = 0; i < 2; ++i) {
      a[i].h[0] = *(const v8bf*)(aptr[i] + kb);
      a[i].h[1] = *(const v8bf*)(aptr[i] + kb + 16);
    }
#pragma unroll
    for (int j = 0; j < 4; ++j) {
      b[j].h[0] = *(const v8bf*)(bptr[j] + kb);
      b[j].h[1] = *(const v8bf*)(bptr[j] + kb + 16);
    }
#pragma unroll
    for (int i = 0; i < 2; ++i)
#pragma unroll
      for (int j = 0; j < 4; ++j)
        acc[i][j] = __builtin_amdgcn_wmma_f32_16x16x32_bf16(
            false, a[i].v, false, b[j].v, (short)0, acc[i][j], false, false);
  }

  // epilogue: C VGPR v holds row (v + 8*hi) of the 16-row sub-tile, col = lane&15
#pragma unroll
  for (int i = 0; i < 2; ++i) {
    const int mbase = m0 + i * 16 + 8 * hi;
#pragma unroll
    for (int j = 0; j < 4; ++j) {
      const int col = n0 + j * 16 + lr;
      const float bcol = bias[col];
#pragma unroll
      for (int v = 0; v < 8; ++v) {
        const int m = mbase + v;
        if (m >= M) continue;
        float val = acc[i][j][v] + bcol;
        size_t mm = MAP_OUT ? (size_t)((m / rows_per) * tok_per + 1 + (m % rows_per))
                            : (size_t)m;
        size_t off = mm * (size_t)N + col;
        if (EPI == 1) {
          out_bf[off] = (bf16)val;
        } else if (EPI == 2) {
          val += (float)add_bf[off];
          out_bf[off] = (bf16)val;
        } else if (EPI == 3) {
          val += add_f32[off];
          out_f32[off] = val;
        } else if (EPI == 4) {
          val = 0.5f * val * (1.0f + erff(val * 0.70710678118f));
          out_bf[off] = (bf16)val;
        }
      }
    }
  }
}

// ---------------- elementwise / reduction kernels ----------------
__global__ __launch_bounds__(256) void cvt_bf16_kernel(const float* __restrict__ s,
                                                       bf16* __restrict__ d, int n) {
  int i = blockIdx.x * 256 + threadIdx.x;
  if (i < n) d[i] = (bf16)s[i];
}

// LayerNorm over D=256, one wave per row, output bf16
__global__ __launch_bounds__(256) void ln_cast_kernel(const float* __restrict__ x,
                                                      const float* __restrict__ g,
                                                      const float* __restrict__ b,
                                                      bf16* __restrict__ o, int rows) {
  int wid = threadIdx.x >> 5, lane = threadIdx.x & 31;
  int row = blockIdx.x * 8 + wid;
  if (row >= rows) return;
  const float* xr = x + (size_t)row * 256;
  float v[8];
  float s = 0.f;
#pragma unroll
  for (int i = 0; i < 8; ++i) { v[i] = xr[i * 32 + lane]; s += v[i]; }
  s = wave_reduce_sum(s);
  float mu = s * (1.0f / 256.0f);
  float s2 = 0.f;
#pragma unroll
  for (int i = 0; i < 8; ++i) { float d = v[i] - mu; s2 += d * d; }
  s2 = wave_reduce_sum(s2);
  float rstd = rsqrtf(s2 * (1.0f / 256.0f) + 1e-5f);
  bf16* orow = o + (size_t)row * 256;
#pragma unroll
  for (int i = 0; i < 8; ++i) {
    int c = i * 32 + lane;
    orow[c] = (bf16)((v[i] - mu) * rstd * g[c] + b[c]);
  }
}

// L2-normalize a bf16 row of length D in place (block per row)
__global__ __launch_bounds__(256) void l2row_kernel(bf16* __restrict__ q, int D) {
  bf16* r = q + (size_t)blockIdx.x * D;
  float s = 0.f;
  for (int c = threadIdx.x; c < D; c += 256) { float v = (float)r[c]; s += v * v; }
  s = block_reduce_sum_256(s);
  float rinv = 1.0f / fmaxf(sqrtf(s), 1e-12f);
  for (int c = threadIdx.x; c < D; c += 256) r[c] = (bf16)((float)r[c] * rinv);
}

// A[row] = scale * dot(qhat[row,:], w_g)
__global__ __launch_bounds__(256) void adot_kernel(const bf16* __restrict__ q,
                                                   const float* __restrict__ wg,
                                                   float* __restrict__ A, int D,
                                                   float scale) {
  const bf16* r = q + (size_t)blockIdx.x * D;
  float s = 0.f;
  for (int c = threadIdx.x; c < D; c += 256) s += (float)r[c] * wg[c];
  s = block_reduce_sum_256(s);
  if (threadIdx.x == 0) A[blockIdx.x] = s * scale;
}

// per-batch 1/max(||A||,eps) over tokens
__global__ __launch_bounds__(256) void anorm_kernel(const float* __restrict__ A,
                                                    float* __restrict__ rinv, int tokens) {
  const float* Ab = A + (size_t)blockIdx.x * tokens;
  float s = 0.f;
  for (int n = threadIdx.x; n < tokens; n += 256) { float v = Ab[n]; s += v * v; }
  s = block_reduce_sum_256(s);
  if (threadIdx.x == 0) rinv[blockIdx.x] = 1.0f / fmaxf(sqrtf(s), 1e-12f);
}

// G[b,d] = sum_n (A[b,n]*rinvA[b]) * qhat[b,n,d]
__global__ __launch_bounds__(256) void gmat_kernel(const bf16* __restrict__ q,
                                                   const float* __restrict__ A,
                                                   const float* __restrict__ rA,
                                                   float* __restrict__ G, int tokens,
                                                   int D) {
  int b = blockIdx.y;
  int d = blockIdx.x * 256 + threadIdx.x;
  float ra = rA[b];
  const bf16* qb = q + (size_t)b * tokens * D + d;
  const float* Ab = A + (size_t)b * tokens;
  float s = 0.f;
  for (int n = 0; n < tokens; ++n) s += Ab[n] * (float)qb[(size_t)n * D];
  G[(size_t)b * D + d] = s * ra;
}

// k[b,n,d] *= G[b,d]  (in place)
__global__ __launch_bounds__(256) void kscale_kernel(bf16* __restrict__ k,
                                                     const float* __restrict__ G,
                                                     int tokens, int D, long long total) {
  long long stride = (long long)gridDim.x * 256;
  for (long long i = (long long)blockIdx.x * 256 + threadIdx.x; i < total; i += stride) {
    int d = (int)(i % D);
    long long row = i / D;
    int b = (int)(row / tokens);
    k[i] = (bf16)((float)k[i] * G[(size_t)b * D + d]);
  }
}

// cls token path: out[b,0,:] = x2[b,0,:] + xn2[b,0,:]
__global__ __launch_bounds__(256) void cls_kernel(const float* __restrict__ x2,
                                                  const bf16* __restrict__ xn2,
                                                  float* __restrict__ out, int tokens) {
  size_t idx = (size_t)blockIdx.x * tokens * 256 + threadIdx.x;
  out[idx] = x2[idx] + (float)xn2[idx];
}

// ---------------- driver ----------------
extern "C" void kernel_launch(void* const* d_in, const int* in_sizes, int n_in,
                              void* d_out, int out_size, void* d_ws, size_t ws_size,
                              hipStream_t stream) {
  (void)in_sizes; (void)n_in; (void)out_size; (void)ws_size;
  const float* x   = (const float*)d_in[0];
  const float* n1w = (const float*)d_in[1];
  const float* n1b = (const float*)d_in[2];
  const float* wq  = (const float*)d_in[3];
  const float* bq  = (const float*)d_in[4];
  const float* wk  = (const float*)d_in[5];
  const float* bk  = (const float*)d_in[6];
  const float* wg  = (const float*)d_in[7];
  const float* wp  = (const float*)d_in[8];
  const float* bp  = (const float*)d_in[9];
  const float* wf  = (const float*)d_in[10];
  const float* bfv = (const float*)d_in[11];
  const float* n2w = (const float*)d_in[12];
  const float* n2b = (const float*)d_in[13];
  const float* w1  = (const float*)d_in[14];
  const float* b1  = (const float*)d_in[15];
  const float* w2  = (const float*)d_in[16];
  const float* b2  = (const float*)d_in[17];
  float* out = (float*)d_out;

  constexpr int Bb = 16, TOK = 3137, BODY = 3136, DIN = 256, DI = 2048, DH = 1024;
  constexpr int M = Bb * TOK;       // 50192
  constexpr int MBODY = Bb * BODY;  // 50176

  char* p = (char*)d_ws;
  auto carve = [&](size_t bytes) -> void* {
    void* r = (void*)p;
    p += (bytes + 255) & ~(size_t)255;
    return r;
  };
  bf16* xn   = (bf16*)carve((size_t)M * DIN * 2);
  bf16* qh   = (bf16*)carve((size_t)M * DI * 2);   // q (later: wp-output)
  bf16* kh   = (bf16*)carve((size_t)M * DI * 2);   // k (later: G*khat)
  float* x2  = (float*)carve((size_t)M * DIN * 4);
  bf16* xn2  = (bf16*)carve((size_t)M * DIN * 2);
  bf16* hdn  = (bf16*)carve((size_t)MBODY * DH * 2);
  float* Av  = (float*)carve((size_t)M * 4);
  float* rA  = (float*)carve((size_t)Bb * 4);
  float* G   = (float*)carve((size_t)Bb * DI * 4);
  bf16* wqb  = (bf16*)carve((size_t)DI * DIN * 2);
  bf16* wkb  = (bf16*)carve((size_t)DI * DIN * 2);
  bf16* wpb  = (bf16*)carve((size_t)DI * DI * 2);
  bf16* wfb  = (bf16*)carve((size_t)DIN * DI * 2);
  bf16* w1b  = (bf16*)carve((size_t)DH * DIN * 2);
  bf16* w2b  = (bf16*)carve((size_t)DIN * DH * 2);

  auto cvt = [&](const float* s, bf16* d, int n) {
    cvt_bf16_kernel<<<(n + 255) / 256, 256, 0, stream>>>(s, d, n);
  };
  cvt(wq, wqb, DI * DIN);
  cvt(wk, wkb, DI * DIN);
  cvt(wp, wpb, DI * DI);
  cvt(wf, wfb, DIN * DI);
  cvt(w1, w1b, DH * DIN);
  cvt(w2, w2b, DIN * DH);

  // 1) LN1 -> xn (bf16)
  ln_cast_kernel<<<(M + 7) / 8, 256, 0, stream>>>(x, n1w, n1b, xn, M);

  // 2) q = xn*wq^T + bq ; k = xn*wk^T + bk (bf16 out)
  {
    dim3 g((M + 127) / 128, DI / 64);
    gemm_bf16_wmma<1, false, false><<<g, 128, 0, stream>>>(
        xn, wqb, bq, nullptr, nullptr, qh, nullptr, M, DI, DIN, BODY, TOK);
    gemm_bf16_wmma<1, false, false><<<g, 128, 0, stream>>>(
        xn, wkb, bk, nullptr, nullptr, kh, nullptr, M, DI, DIN, BODY, TOK);
  }

  // 3) L2 row-normalize q, k in place
  l2row_kernel<<<M, 256, 0, stream>>>(qh, DI);
  l2row_kernel<<<M, 256, 0, stream>>>(kh, DI);

  // 4) A = scale * qhat @ w_g ; per-batch normalize ; G = sum_n A'*qhat
  adot_kernel<<<M, 256, 0, stream>>>(qh, wg, Av, DI, 0.0625f);
  anorm_kernel<<<Bb, 256, 0, stream>>>(Av, rA, TOK);
  {
    dim3 g(DI / 256, Bb);
    gmat_kernel<<<g, 256, 0, stream>>>(qh, Av, rA, G, TOK, DI);
  }

  // 5) k <- G * khat (in place)
  kscale_kernel<<<8192, 256, 0, stream>>>(kh, G, TOK, DI, (long long)M * DI);

  // 6) out = (G*khat)@wp^T + bp + qhat  -> overwrite qh (bf16)
  {
    dim3 g((M + 127) / 128, DI / 64);
    gemm_bf16_wmma<2, false, false><<<g, 128, 0, stream>>>(
        kh, wpb, bp, qh, nullptr, qh, nullptr, M, DI, DI, BODY, TOK);
  }

  // 7) x2 = x + out@wf^T + bf  (f32)
  {
    dim3 g((M + 127) / 128, DIN / 64);
    gemm_bf16_wmma<3, false, false><<<g, 128, 0, stream>>>(
        qh, wfb, bfv, nullptr, x, nullptr, x2, M, DIN, DI, BODY, TOK);
  }

  // 8) LN2 -> xn2 (bf16)
  ln_cast_kernel<<<(M + 7) / 8, 256, 0, stream>>>(x2, n2w, n2b, xn2, M);

  // 9) hdn = gelu(xn2[body]@w1^T + b1) (bf16, body rows gathered)
  {
    dim3 g(MBODY / 128, DH / 64);
    gemm_bf16_wmma<4, true, false><<<g, 128, 0, stream>>>(
        xn2, w1b, b1, nullptr, nullptr, hdn, nullptr, MBODY, DH, DIN, BODY, TOK);
  }

  // 10) out[body] = x2[body] + hdn@w2^T + b2 (f32, scatter to body rows)
  {
    dim3 g(MBODY / 128, DIN / 64);
    gemm_bf16_wmma<3, false, true><<<g, 128, 0, stream>>>(
        hdn, w2b, b2, nullptr, x2, nullptr, out, MBODY, DIN, DH, BODY, TOK);
  }

  // 11) out[cls] = x2[cls] + xn2[cls]
  cls_kernel<<<Bb, 256, 0, stream>>>(x2, xn2, out, TOK);
}
